// RBF_SVM_79869211836958
// MI455X (gfx1250) — matrix operand order repaired and checked by reading, not compiled, not run on hardware
//
#include <hip/hip_runtime.h>
#include <hip/hip_bf16.h>
#include <stdint.h>

typedef __attribute__((ext_vector_type(16))) __bf16 bf16x16;
typedef __attribute__((ext_vector_type(8)))  float  f32x8;

constexpr int N = 4096;
constexpr int M = 8192;
constexpr int D = 512;

constexpr size_t XBF_BYTES  = (size_t)N * D * 2;   // 4 MB
constexpr size_t SVBF_BYTES = (size_t)M * D * 2;   // 8 MB

static __device__ __forceinline__ uint32_t pack_bf16x2(float a, float b) {
    __hip_bfloat16 ha = __float2bfloat16(a);
    __hip_bfloat16 hb = __float2bfloat16(b);
    uint16_t ua, ub;
    __builtin_memcpy(&ua, &ha, 2);
    __builtin_memcpy(&ub, &hb, 2);
    return (uint32_t)ua | ((uint32_t)ub << 16);
}

// One wave32 per row: convert a f32 row (D=512) to bf16 and compute its exact
// f32 squared L2 norm. 16 elements per lane as 4x float4 loads.
__global__ __launch_bounds__(256)
void prep_rows(const float* __restrict__ src, uint32_t* __restrict__ dst_bf,
               float* __restrict__ norms, int rows) {
    int gwave = (int)((blockIdx.x * blockDim.x + threadIdx.x) >> 5);
    int lane  = (int)(threadIdx.x & 31);
    if (gwave >= rows) return;
    const float4* r = (const float4*)(src + (size_t)gwave * D);
    uint32_t* o = dst_bf + (size_t)gwave * (D / 2);
    float ss = 0.f;
#pragma unroll
    for (int i = 0; i < 4; ++i) {
        int q = lane + i * 32;           // float4 index within row (0..127)
        float4 v = r[q];
        ss += v.x * v.x + v.y * v.y + v.z * v.z + v.w * v.w;
        o[q * 2 + 0] = pack_bf16x2(v.x, v.y);
        o[q * 2 + 1] = pack_bf16x2(v.z, v.w);
    }
#pragma unroll
    for (int off = 16; off > 0; off >>= 1) ss += __shfl_xor(ss, off, 32);
    if (lane == 0) norms[gwave] = ss;
}

#define WAVES 8
#define NREP  2   // 16-row n-subtiles per wave (workgroup owns 32 rows)
#define MREP  4   // 16-col m-subtiles per wave (wave owns 64 cols / iter)

// Block = 256 threads (8 waves) owns 32 rows of X and sweeps all of M.
// Wave w covers m columns [m0 + w*64, m0 + w*64 + 64) per outer iteration.
// Per k-step: 4 A-loads + 8 B-loads (b128) feed 8 WMMAs (2x4 accumulators).
__global__ __launch_bounds__(256)
void rbf_main(const uint16_t* __restrict__ Xbf,
              const uint16_t* __restrict__ SVbf,
              const float* __restrict__ x2,
              const float* __restrict__ sv2,
              const float* __restrict__ DC,
              const float* __restrict__ ICp,
              const float* __restrict__ gammap,
              float* __restrict__ out) {
    __shared__ float red[WAVES][16 * NREP][16];

    const int tid  = (int)threadIdx.x;
    const int wave = tid >> 5;
    const int lane = tid & 31;
    const int h    = lane >> 4;   // lane half (0/1)
    const int lr   = lane & 15;   // lane row/col within half
    const int n0   = (int)blockIdx.x * (16 * NREP);

    const float gamma = gammap[0];

    // x2 for the rows this lane touches (C/D row = j + 8*h within each n-tile)
    float x2l[NREP][8];
#pragma unroll
    for (int r = 0; r < NREP; ++r)
#pragma unroll
        for (int j = 0; j < 8; ++j) x2l[r][j] = x2[n0 + r * 16 + j + 8 * h];

    float psum[NREP][8];
#pragma unroll
    for (int r = 0; r < NREP; ++r)
#pragma unroll
        for (int j = 0; j < 8; ++j) psum[r][j] = 0.f;

    // A fragment bases: row n0 + r*16 + lr, K chunk starts at 8*h (ISA layout)
    const uint16_t* arow_base[NREP];
#pragma unroll
    for (int r = 0; r < NREP; ++r)
        arow_base[r] = Xbf + (size_t)(n0 + r * 16 + lr) * D + 8 * h;

    for (int m0 = wave * (16 * MREP); m0 < M; m0 += WAVES * 16 * MREP) {
        f32x8 acc[NREP][MREP];
#pragma unroll
        for (int r = 0; r < NREP; ++r)
#pragma unroll
            for (int t = 0; t < MREP; ++t)
                acc[r][t] = (f32x8){0.f,0.f,0.f,0.f,0.f,0.f,0.f,0.f};

#pragma unroll 2
        for (int k0 = 0; k0 < D; k0 += 32) {
            union { bf16x16 v; uint4 u[2]; } A[NREP];
#pragma unroll
            for (int r = 0; r < NREP; ++r) {
                const uint16_t* ar = arow_base[r] + k0;
                A[r].u[0] = *(const uint4*)(ar);        // K = k0+8h    .. +7
                A[r].u[1] = *(const uint4*)(ar + 16);   // K = k0+16+8h .. +7
            }
#pragma unroll
            for (int t = 0; t < MREP; ++t) {
                union { bf16x16 v; uint4 u[2]; } B;
                // B col = lr -> SV row m0+t*16+lr; lane holds K = k0+8h..+7, k0+16+8h..+7
                const uint16_t* br = SVbf + (size_t)(m0 + t * 16 + lr) * D + k0 + 8 * h;
                B.u[0] = ((const uint4*)br)[0];
                B.u[1] = ((const uint4*)br)[2];
#pragma unroll
                for (int r = 0; r < NREP; ++r)
                    acc[r][t] = __builtin_amdgcn_wmma_f32_16x16x32_bf16(
                        /*neg_a=*/false, A[r].v, /*neg_b=*/false, B.v,
                        /*c_mod=*/(short)0, acc[r][t],
                        /*reuse_a=*/false, /*reuse_b=*/false);
            }
        }

        // Fused RBF epilogue: d2 -> exp -> *DC, accumulate per-lane partials.
        // C/D layout: VGPR j holds row (j + 8h), column lr.
#pragma unroll
        for (int t = 0; t < MREP; ++t) {
            const int mcol = m0 + t * 16 + lr;
            const float s2 = sv2[mcol];
            const float dc = DC[mcol];
#pragma unroll
            for (int r = 0; r < NREP; ++r)
#pragma unroll
                for (int j = 0; j < 8; ++j) {
                    float d2 = (x2l[r][j] + s2) - 2.0f * acc[r][t][j];
                    d2 = fmaxf(d2, 0.0f);
                    psum[r][j] += __expf(-gamma * d2) * dc;
                }
        }
    }

    // Cross-lane / cross-wave reduction: each (wave, local-row, col-lane) slot
    // written exactly once -> fully deterministic (no float atomics).
#pragma unroll
    for (int r = 0; r < NREP; ++r)
#pragma unroll
        for (int j = 0; j < 8; ++j)
            red[wave][r * 16 + j + 8 * h][lr] = psum[r][j];
    __syncthreads();

    if (tid < 16 * NREP) {
        float s = 0.f;
        for (int w = 0; w < WAVES; ++w)
#pragma unroll
            for (int c = 0; c < 16; ++c) s += red[w][tid][c];
        out[n0 + tid] = s + ICp[0];
    }
}

extern "C" void kernel_launch(void* const* d_in, const int* in_sizes, int n_in,
                              void* d_out, int out_size, void* d_ws, size_t ws_size,
                              hipStream_t stream) {
    const float* X     = (const float*)d_in[0];   // [N, D]
    const float* SV    = (const float*)d_in[1];   // [M, D]
    const float* DC    = (const float*)d_in[2];   // [M]
    const float* IC    = (const float*)d_in[3];   // [1]
    const float* gamma = (const float*)d_in[4];   // [1]
    float* out = (float*)d_out;

    char* ws = (char*)d_ws;
    uint32_t* Xbf  = (uint32_t*)ws;
    uint32_t* SVbf = (uint32_t*)(ws + XBF_BYTES);
    float*    x2   = (float*)(ws + XBF_BYTES + SVBF_BYTES);
    float*    sv2  = x2 + N;

    // bf16 conversion + exact f32 row norms (wave32 per row, 8 waves/block)
    prep_rows<<<N / 8, 256, 0, stream>>>(X,  Xbf,  x2,  N);
    prep_rows<<<M / 8, 256, 0, stream>>>(SV, SVbf, sv2, M);

    // Main fused WMMA GEMM + RBF + weighted reduction. 128 blocks of 32 rows.
    rbf_main<<<N / (16 * NREP), 256, 0, stream>>>((const uint16_t*)Xbf,
                                                  (const uint16_t*)SVbf,
                                                  x2, sv2, DC, IC, gamma, out);
}
